// FloodML_Conv_LSTM_88553635709274
// MI455X (gfx1250) — matrix operand order repaired and checked by reading, not compiled, not run on hardware
//
#include <hip/hip_runtime.h>

// ---------------------------------------------------------------------------
// ConvLSTM for MI455X (gfx1250): implicit-GEMM 3x3 convs on the WMMA pipe.
// gate_preact = conv(x_t;Wx) + conv(h_prev;Wh) + bias  as one GEMM with
// K = 18 taps * 32 in-channels; each tap is one v_wmma_f32_16x16x32_bf16.
// K-loop is software-pipelined (2-stage) so tap t+1's A/B fragment loads
// overlap tap t's WMMAs instead of stalling on s_wait_loadcnt 0.
// ---------------------------------------------------------------------------

typedef __attribute__((ext_vector_type(16))) __bf16 v16bf;
typedef __attribute__((ext_vector_type(8)))  float  v8f;
typedef __attribute__((ext_vector_type(4)))  float  f4;

union Frag { v16bf v; f4 q[2]; };

__device__ __forceinline__ unsigned short f2bf(float f) {
    unsigned int u = __float_as_uint(f);
    u += 0x7FFFu + ((u >> 16) & 1u);          // round-to-nearest-even
    return (unsigned short)(u >> 16);
}
__device__ __forceinline__ float sigm(float x) { return 1.f / (1.f + __expf(-x)); }
__device__ __forceinline__ float tanha(float x) {
    x = fminf(fmaxf(x, -15.f), 15.f);
    float e = __expf(2.f * x);
    return (e - 1.f) / (e + 1.f);
}

// Pre-transform weights: Wt[tap(18)][o(128)][cin(32)] bf16, taps 0-8 = Wx(ky*3+kx),
// taps 9-17 = Wh. Source layout: W[4,32,32,3,3] -> (o*32+cin)*9 + tapk.
__global__ void build_weights(const float* __restrict__ Wx,
                              const float* __restrict__ Wh,
                              unsigned short* __restrict__ Wt) {
    int idx = blockIdx.x * 256 + threadIdx.x;
    if (idx >= 18 * 128 * 32) return;
    int cin = idx & 31;
    int o   = (idx >> 5) & 127;
    int tap = idx >> 12;
    const float* W = (tap < 9) ? Wx : Wh;
    int tk = (tap < 9) ? tap : tap - 9;
    Wt[idx] = f2bf(W[(o * 32 + cin) * 9 + tk]);
}

// One workgroup per (batch b, row y): 64 px x 128 gate-channels.
__global__ void __launch_bounds__(256) convlstm_step(
    const float* __restrict__ x, int t,
    const unsigned short* __restrict__ Wt,
    const float* __restrict__ bx, const float* __restrict__ bh,
    const float* __restrict__ hprev,
    float* __restrict__ cstate, float* __restrict__ hnext)
{
    __shared__ __align__(16) unsigned short xpatch[3 * 66 * 32]; // [row][px][c] bf16
    __shared__ __align__(16) unsigned short hpatch[3 * 66 * 32];
    __shared__ float preact[64 * 132];                           // [px][o], padded stride

    const int tid = threadIdx.x;
    const int y = blockIdx.x & 63;
    const int b = blockIdx.x >> 6;

    const float* xsrc = x + ((size_t)(b * 16 + t) << 17);  // * 32*4096
    const float* hsrc = hprev + ((size_t)b << 17);

    // ---- stage halo patches (rows y-1..y+1, px -1..64), f32 -> bf16 ----
    for (int s = tid; s < 3 * 32 * 66; s += 256) {
        int px = s % 66;
        int rc = s / 66;
        int c  = rc & 31;
        int r  = rc >> 5;
        int gy = y + r - 1, gx = px - 1;
        bool in = ((unsigned)gy < 64u) & ((unsigned)gx < 64u);
        int go = ((c << 6) + gy) * 64 + gx;
        float xv = in ? xsrc[go] : 0.f;
        float hv = in ? hsrc[go] : 0.f;
        int li = (r * 66 + px) * 32 + c;
        xpatch[li] = f2bf(xv);
        hpatch[li] = f2bf(hv);
    }
    __syncthreads();

    // ---- WMMA GEMM: wave -> 16 px (M-tile) x 64 ch (4 N-tiles) ----
    const int lane = tid & 31;
    const int wv   = tid >> 5;
    const int mt   = wv & 3;
    const int nbase = (wv >> 2) << 6;
    const int m    = lane & 15;
    const int hi   = lane >> 4;       // lane half
    const int px0  = mt << 4;
    const int cbA  = hi << 4;         // A K-layout: 0 | 16  (K 0-15 vs 16-31)
    const int cbB  = hi << 4;         // B K-layout: 0 | 16  (K 0-15 vs 16-31)

    v8f acc[4];
#pragma unroll
    for (int j = 0; j < 4; ++j) {
        int o = nbase + (j << 4) + m;
        float bias = bx[o] + bh[o];
#pragma unroll
        for (int r = 0; r < 8; ++r) acc[j][r] = bias;
    }

    Frag Af[2];
    Frag Bf[2][4];

    // prologue: load fragments for tap 0 into buffer 0
    {
        const unsigned short* ap = xpatch + ((px0 + m) << 5) + cbA;  // dy=0,dx=0
        Af[0].q[0] = *(const f4*)(ap);
        Af[0].q[1] = *(const f4*)(ap + 8);
#pragma unroll
        for (int j = 0; j < 4; ++j) {
            const unsigned short* wp = Wt + ((nbase + (j << 4) + m) << 5) + cbB;
            Bf[0][j].q[0] = *(const f4*)(wp);
            Bf[0][j].q[1] = *(const f4*)(wp + 8);
        }
    }

#pragma unroll
    for (int tap = 0; tap < 18; ++tap) {
        const int cur = tap & 1;
        const int nxt = cur ^ 1;
        // prefetch tap+1 fragments into the other buffer (overlaps WMMAs below)
        if (tap < 17) {
            const int tn = tap + 1;
            const unsigned short* patch = (tn < 9) ? xpatch : hpatch;
            const int tk = (tn < 9) ? tn : tn - 9;
            const int dy = tk / 3, dx = tk % 3;
            const unsigned short* ap = patch + (((dy * 66) + px0 + m + dx) << 5) + cbA;
            Af[nxt].q[0] = *(const f4*)(ap);
            Af[nxt].q[1] = *(const f4*)(ap + 8);
            const unsigned short* wrow = Wt + (tn << 12);
#pragma unroll
            for (int j = 0; j < 4; ++j) {
                const unsigned short* wp = wrow + ((nbase + (j << 4) + m) << 5) + cbB;
                Bf[nxt][j].q[0] = *(const f4*)(wp);
                Bf[nxt][j].q[1] = *(const f4*)(wp + 8);
            }
        }
#pragma unroll
        for (int j = 0; j < 4; ++j) {
            acc[j] = __builtin_amdgcn_wmma_f32_16x16x32_bf16(
                false, Af[cur].v, false, Bf[cur][j].v, (short)0, acc[j], false, false);
        }
    }

    // ---- scatter D tiles to LDS: preact[pixel][gate-channel] ----
#pragma unroll
    for (int j = 0; j < 4; ++j) {
        int n = nbase + (j << 4) + m;
#pragma unroll
        for (int r = 0; r < 8; ++r) {
            int p = px0 + r + (hi << 3);          // M = r + 8*lane_half
            preact[p * 132 + n] = acc[j][r];
        }
    }
    __syncthreads();

    // ---- gates + state update (64 px x 32 ch), c in-place, h double-buffered ----
    for (int s = tid; s < 2048; s += 256) {
        int px = s & 63;
        int c  = s >> 6;
        const float* g = preact + px * 132;
        float zi = g[c];
        float zf = g[c + 32];
        float zo = g[c + 64];
        float zg = g[c + 96];
        size_t off = (((size_t)(b * 32 + c) << 6) + y) * 64 + px;
        float cp = cstate[off];
        float iv = sigm(zi), fv = sigm(zf), ov = sigm(zo), gv = tanha(zg);
        float cn = fv * cp + iv * gv;
        cstate[off] = cn;
        hnext[off]  = ov * tanha(cn);
    }
}

extern "C" void kernel_launch(void* const* d_in, const int* in_sizes, int n_in,
                              void* d_out, int out_size, void* d_ws, size_t ws_size,
                              hipStream_t stream) {
    (void)in_sizes; (void)n_in; (void)out_size; (void)ws_size;
    const float* x    = (const float*)d_in[0];
    const float* c0   = (const float*)d_in[1];
    const float* h0in = (const float*)d_in[2];
    const float* Wx   = (const float*)d_in[3];
    const float* bx   = (const float*)d_in[4];
    const float* Wh   = (const float*)d_in[5];
    const float* bh   = (const float*)d_in[6];

    const size_t STATE_BYTES = (size_t)8 * 32 * 64 * 64 * 4;   // 4 MB
    char* wsb = (char*)d_ws;
    unsigned short* Wt = (unsigned short*)wsb;                  // 147456 B
    float* cst = (float*)(wsb + 147456);
    float* hA  = (float*)(wsb + 147456 + STATE_BYTES);
    float* hB  = (float*)(wsb + 147456 + 2 * STATE_BYTES);

    hipMemcpyAsync(cst, c0,   STATE_BYTES, hipMemcpyDeviceToDevice, stream);
    hipMemcpyAsync(hA,  h0in, STATE_BYTES, hipMemcpyDeviceToDevice, stream);

    build_weights<<<288, 256, 0, stream>>>(Wx, Wh, Wt);

    for (int t = 0; t < 16; ++t) {
        convlstm_step<<<512, 256, 0, stream>>>(x, t, Wt, bx, bh, hA, cst, hB);
        float* tmp = hA; hA = hB; hB = tmp;
    }
    hipMemcpyAsync(d_out, hA, STATE_BYTES, hipMemcpyDeviceToDevice, stream);
}
